// NeighborEmbedding_19808389169521
// MI455X (gfx1250) — compile-verified, matrix operand order checked by reading, not run on hardware
//
#include <hip/hip_runtime.h>
#include <hip/hip_fp16.h>

#define N_NODES   100000
#define N_EDGES   1600000
#define HIDDEN    128
#define NUM_RBF   50
#define KPAD      64
#define CUTOFF_F  5.0f
#define PI_F      3.14159265358979f

#define EPB 32   // edges per block   (N_EDGES % 32 == 0 -> no tail)
#define NPB 32   // nodes per block   (N_NODES % 32 == 0 -> no tail)

typedef __attribute__((ext_vector_type(16))) _Float16     v16h;
typedef __attribute__((ext_vector_type(8)))  float        v8f;
typedef __attribute__((ext_vector_type(4)))  unsigned int v4u;
typedef __attribute__((ext_vector_type(8)))  int          v8i;
typedef __attribute__((ext_vector_type(4)))  int          v4i;

#if defined(__has_builtin)
#  if __has_builtin(__builtin_amdgcn_tensor_load_to_lds)
#    define HAVE_TDM 1
#  endif
#endif
#ifndef HAVE_TDM
#  define HAVE_TDM 0
#endif

#if defined(__has_builtin)
#  if __has_builtin(__builtin_amdgcn_s_wait_tensorcnt)
#    define WAIT_TENSORCNT0() __builtin_amdgcn_s_wait_tensorcnt(0)
#  else
#    define WAIT_TENSORCNT0() asm volatile("s_wait_tensorcnt 0x0" ::: "memory")
#  endif
#else
#  define WAIT_TENSORCNT0() asm volatile("s_wait_tensorcnt 0x0" ::: "memory")
#endif

// Low 32 bits of a flat shared pointer == LDS byte offset (flat->LDS truncates
// to addr[31:0] per ISA §10.2 aperture mapping).
__device__ __forceinline__ unsigned lds_addr_of(const void* p) {
    return (unsigned)(unsigned long long)p;
}

// K index held by ext-vector element i for 16-bit WMMA A/B operands (wave32).
// A: lanes 0-15 -> M=lane, lanes 16-31 -> M=lane-16 (upper K half-groups)
// B: lanes 0-15 -> N=lane, lanes 16-31 -> N=lane-16
__device__ __forceinline__ int wmma_k(int i, int half) {
    return ((i >> 3) << 4) + (half << 3) + (i & 7);
}

// ---------------------------------------------------------------------------
// Kernel 1: fused per-edge pipeline.
//   W = (edge_attr @ dp_w.T + dp_b) * cutoff(r);  msg = W * emb[z[dst]]
//   agg[src] += msg   (f32 atomics)
// Block = 256 threads (8 waves). Block tile: 32 edges x 128 hidden.
// TDM stages the f32 edge_attr tile; one conversion pass builds a zero-padded
// f16 tile so the WMMA operand build is branch-free f16 LDS reads.
// ---------------------------------------------------------------------------
__global__ __launch_bounds__(256) void edge_msg_kernel(
    const float* __restrict__ edge_attr,    // [E, 50]
    const float* __restrict__ edge_weight,  // [E]
    const int*   __restrict__ edge_index,   // [2, E] row0=src(scatter) row1=dst(gather)
    const int*   __restrict__ z,            // [N]
    const float* __restrict__ emb,          // [95, 128]
    const float* __restrict__ dp_w,         // [128, 50]
    const float* __restrict__ dp_b,         // [128]
    float*       __restrict__ agg)          // [N, 128] (pre-zeroed)
{
    __shared__ _Float16 sA[EPB][KPAD];      // f16 tile, zero-padded K (WMMA A source)
    __shared__ float    sEmb[EPB][HIDDEN];  // gathered emb[z[dst]] rows
    __shared__ float    sC[EPB];            // cosine cutoff per edge
    __shared__ int      sSrc[EPB];          // scatter row per edge
    __shared__ int      sRow[EPB];          // emb row base (z[dst]*HIDDEN)
#if HAVE_TDM
    __shared__ float    sRaw[EPB * NUM_RBF]; // compact f32 TDM destination
#endif

    const int tid   = threadIdx.x;
    const int lane  = tid & 31;
    const int wave  = tid >> 5;             // 0..7
    const int half  = lane >> 4;            // 0 | 1
    const int nl    = lane & 15;
    const int eBase = blockIdx.x * EPB;

#if HAVE_TDM
    if (wave == 0) {
        // Per-edge metadata handled by wave 0's 32 lanes.
        int e = eBase + lane;
        float r = edge_weight[e];
        sC[lane]   = (r < CUTOFF_F) ? 0.5f * (__cosf(r * (PI_F / CUTOFF_F)) + 1.0f) : 0.0f;
        sSrc[lane] = edge_index[e];                       // row 0: scatter target
        sRow[lane] = z[edge_index[N_EDGES + e]] * HIDDEN; // row 1 -> emb row base

        // TDM: 2-D tile [32 rows x 50 f32] of tensor edge_attr[E][50] -> LDS.
        unsigned long long ga =
            (unsigned long long)edge_attr + (unsigned long long)eBase * (NUM_RBF * 4);
        v4u g0;
        g0[0] = 1u;                                       // count=1 (valid descriptor)
        g0[1] = lds_addr_of(&sRaw[0]);                    // lds_addr
        g0[2] = (unsigned)ga;                             // global_addr[31:0]
        g0[3] = (unsigned)((ga >> 32) & 0x1FFFFFFull) | (2u << 30); // addr[56:32] | type=2
        v8i g1;
        g1[0] = (int)(2u << 16);                          // data_size=4B
        g1[1] = (int)((unsigned)NUM_RBF << 16);           // tensor_dim0 = 50
        g1[2] = (int)(((unsigned)N_EDGES & 0xFFFFu) << 16);          // tensor_dim1 lo16
        g1[3] = (int)((((unsigned)N_EDGES >> 16) & 0xFFFFu)          // tensor_dim1 hi16
                      | ((unsigned)NUM_RBF << 16));                  // tile_dim0 = 50
        g1[4] = (int)EPB;                                 // tile_dim1 = 32, tile_dim2 = 0
        g1[5] = (int)NUM_RBF;                             // tensor_dim0_stride = 50
        g1[6] = 0;
        g1[7] = 0;
        v4i gz4 = {0, 0, 0, 0};
        v8i gz8 = {0, 0, 0, 0, 0, 0, 0, 0};
        __builtin_amdgcn_tensor_load_to_lds(g0, g1, gz4, gz4, gz8, 0);
        WAIT_TENSORCNT0();
    }
    __syncthreads();

    // One-pass f32 -> f16 conversion with zero padding (each element once).
    for (int idx = tid; idx < EPB * KPAD; idx += 256) {   // 8 uniform iters
        int r = idx >> 6, c = idx & 63;
        float v = sRaw[r * NUM_RBF + (c < NUM_RBF ? c : 0)];  // clamped: no branch
        sA[r][c] = (c < NUM_RBF) ? (_Float16)v : (_Float16)0.0f;
    }
#else
    if (tid < EPB) {
        int e = eBase + tid;
        float r = edge_weight[e];
        sC[tid]   = (r < CUTOFF_F) ? 0.5f * (__cosf(r * (PI_F / CUTOFF_F)) + 1.0f) : 0.0f;
        sSrc[tid] = edge_index[e];
        sRow[tid] = z[edge_index[N_EDGES + e]] * HIDDEN;
    }
    __syncthreads();   // sRow ready for gather below; sA filled next
    for (int idx = tid; idx < EPB * KPAD; idx += 256) {
        int r = idx >> 6, c = idx & 63;
        float v = (c < NUM_RBF)
                ? __builtin_nontemporal_load(&edge_attr[(eBase + r) * NUM_RBF + c])
                : 0.0f;
        sA[r][c] = (_Float16)v;
    }
#endif

    // Gather neighbor-type embedding rows (32 x 128 f32, L2-hot table).
    for (int idx = tid; idx < EPB * HIDDEN; idx += 256) { // 16 uniform iters
        int r = idx >> 7, c = idx & 127;
        sEmb[r][c] = emb[sRow[r] + c];
    }
    __syncthreads();

    // B operands (dp_w cols, L2-resident). Tail handled by clamped load + select.
    const int n = (wave << 4) + nl;                       // global hidden column
    v16h b0, b1;
    #pragma unroll
    for (int i = 0; i < 16; ++i) {
        int k0  = wmma_k(i, half);                        // 0..31 (< 50 always)
        int k1  = 32 + k0;                                // 32..63
        int k1c = (k1 < NUM_RBF) ? k1 : (NUM_RBF - 1);    // clamp: always in bounds
        b0[i] = (_Float16)dp_w[n * NUM_RBF + k0];
        float t = dp_w[n * NUM_RBF + k1c];                // unconditional load
        b1[i] = (k1 < NUM_RBF) ? (_Float16)t : (_Float16)0.0f;  // v_cndmask
    }

    v8f acc[2] = {{}, {}};
    #pragma unroll
    for (int mt = 0; mt < 2; ++mt) {
        const int mrow = (mt << 4) + nl;
        v16h a0, a1;
        #pragma unroll
        for (int i = 0; i < 16; ++i) {
            int k0 = wmma_k(i, half);
            a0[i] = sA[mrow][k0];                         // branch-free f16 reads
            a1[i] = sA[mrow][32 + k0];                    // zeros from padding
        }
        acc[mt] = __builtin_amdgcn_wmma_f32_16x16x32_f16(false, a0, false, b0,
                                                         (short)0, acc[mt], false, false);
        acc[mt] = __builtin_amdgcn_wmma_f32_16x16x32_f16(false, a1, false, b1,
                                                         (short)0, acc[mt], false, false);
    }

    // Epilogue: bias, cutoff, embedding gate, scatter-add (branch-free; no tail).
    const float bias = dp_b[n];
    #pragma unroll
    for (int mt = 0; mt < 2; ++mt) {
        #pragma unroll
        for (int r = 0; r < 8; ++r) {
            int m = (mt << 4) + r + (half << 3);          // edge row within tile
            float val = (acc[mt][r] + bias) * sC[m] * sEmb[m][n];
            atomicAdd(&agg[sSrc[m] * HIDDEN + n], val);
        }
    }
}

// ---------------------------------------------------------------------------
// Kernel 2: out = concat(node_feat, agg) @ comb_w.T + comb_b
// Block = 256 threads (8 waves). Block tile: 32 nodes x 128 out cols.
// Two parallel TDM loads (wave0: node_feat, wave1: agg) build the f32 concat
// tile via TDM row padding (128 DW data + 128 DW pad -> row stride 256 f32);
// one conversion pass produces the f16 tile consumed by 16 WMMAs/wave.
// ---------------------------------------------------------------------------
__global__ __launch_bounds__(256) void combine_kernel(
    const float* __restrict__ node_feat,  // [N, 128]
    const float* __restrict__ agg,        // [N, 128]
    const float* __restrict__ comb_w,     // [128, 256]
    const float* __restrict__ comb_b,     // [128]
    float*       __restrict__ out)        // [N, 128]
{
    __shared__ _Float16 sInH[NPB][2 * HIDDEN];           // f16 concat tile (16 KB)
#if HAVE_TDM
    __shared__ float    sIn[NPB][2 * HIDDEN];            // f32 TDM destination (32 KB)
#endif

    const int tid   = threadIdx.x;
    const int lane  = tid & 31;
    const int wave  = tid >> 5;
    const int half  = lane >> 4;
    const int nl    = lane & 15;
    const int mBase = blockIdx.x * NPB;

#if HAVE_TDM
    if (wave < 2) {
        const float* src = (wave == 0) ? node_feat : agg;
        unsigned long long ga =
            (unsigned long long)src + (unsigned long long)mBase * (HIDDEN * 4);
        v4u g0;
        g0[0] = 1u;                                              // count=1
        g0[1] = lds_addr_of(&sIn[0][0]) + (unsigned)wave * (HIDDEN * 4); // col 0 / col 128
        g0[2] = (unsigned)ga;
        g0[3] = (unsigned)((ga >> 32) & 0x1FFFFFFull) | (2u << 30);
        v8i g1;
        // data_size=4B | pad_enable | pad_interval=128DW(code 6) | pad_amount=128DW(code 127)
        g1[0] = (int)((2u << 16) | (1u << 20) | (6u << 22) | (127u << 25));
        g1[1] = (int)((unsigned)HIDDEN << 16);                   // tensor_dim0 = 128
        g1[2] = (int)(((unsigned)N_NODES & 0xFFFFu) << 16);      // tensor_dim1 lo16
        g1[3] = (int)((((unsigned)N_NODES >> 16) & 0xFFFFu)      // tensor_dim1 hi16
                      | ((unsigned)HIDDEN << 16));               // tile_dim0 = 128
        g1[4] = (int)NPB;                                        // tile_dim1 = 32
        g1[5] = (int)HIDDEN;                                     // tensor_dim0_stride = 128
        g1[6] = 0;
        g1[7] = 0;
        v4i gz4 = {0, 0, 0, 0};
        v8i gz8 = {0, 0, 0, 0, 0, 0, 0, 0};
        __builtin_amdgcn_tensor_load_to_lds(g0, g1, gz4, gz4, gz8, 0);
        WAIT_TENSORCNT0();
    }
    __syncthreads();

    // One-pass f32 -> f16 conversion, pairwise (ds_load_b64 + v_cvt_pk + b32 store).
    {
        const float* pIn = &sIn[0][0];
        _Float16*    pH  = &sInH[0][0];
        for (int idx = tid; idx < NPB * HIDDEN; idx += 256) {    // 16 uniform iters
            int b = idx << 1;
            pH[b]     = (_Float16)pIn[b];
            pH[b + 1] = (_Float16)pIn[b + 1];
        }
    }
    __syncthreads();
#else
    for (int idx = tid; idx < NPB * 2 * HIDDEN; idx += 256) {    // 32 uniform iters
        int r = idx >> 8, c = idx & 255;
        int node = mBase + r;
        float v = (c < HIDDEN)
                ? __builtin_nontemporal_load(&node_feat[node * HIDDEN + c])
                : __builtin_nontemporal_load(&agg[node * HIDDEN + (c - HIDDEN)]);
        sInH[r][c] = (_Float16)v;
    }
    __syncthreads();
#endif

    const int n = (wave << 4) + nl;                      // output column
    v8f acc[2] = {{}, {}};
    #pragma unroll
    for (int kb = 0; kb < 8; ++kb) {
        v16h b;
        #pragma unroll
        for (int i = 0; i < 16; ++i) {
            int k = (kb << 5) + wmma_k(i, half);
            b[i] = (_Float16)comb_w[n * (2 * HIDDEN) + k];   // L2-resident weights
        }
        #pragma unroll
        for (int mt = 0; mt < 2; ++mt) {
            v16h a;
            #pragma unroll
            for (int i = 0; i < 16; ++i) {
                int k = (kb << 5) + wmma_k(i, half);
                a[i] = sInH[(mt << 4) + nl][k];              // branch-free f16 reads
            }
            acc[mt] = __builtin_amdgcn_wmma_f32_16x16x32_f16(false, a, false, b,
                                                             (short)0, acc[mt], false, false);
        }
    }

    const float bias = comb_b[n];
    #pragma unroll
    for (int mt = 0; mt < 2; ++mt) {
        #pragma unroll
        for (int r = 0; r < 8; ++r) {
            int node = mBase + (mt << 4) + r + (half << 3);
            __builtin_nontemporal_store(acc[mt][r] + bias, &out[node * HIDDEN + n]);
        }
    }
}

// ---------------------------------------------------------------------------
// Inputs (setup_inputs order): z, node_feat, edge_index, edge_weight,
// edge_attr, emb, dp_w, dp_b, comb_w, comb_b. Output: [N, 128] f32.
// d_ws: agg accumulator [N,128] f32 (51.2 MB).
// ---------------------------------------------------------------------------
extern "C" void kernel_launch(void* const* d_in, const int* in_sizes, int n_in,
                              void* d_out, int out_size, void* d_ws, size_t ws_size,
                              hipStream_t stream) {
    const int*   z           = (const int*)  d_in[0];
    const float* node_feat   = (const float*)d_in[1];
    const int*   edge_index  = (const int*)  d_in[2];
    const float* edge_weight = (const float*)d_in[3];
    const float* edge_attr   = (const float*)d_in[4];
    const float* emb         = (const float*)d_in[5];
    const float* dp_w        = (const float*)d_in[6];
    const float* dp_b        = (const float*)d_in[7];
    const float* comb_w      = (const float*)d_in[8];
    const float* comb_b      = (const float*)d_in[9];
    float* out = (float*)d_out;
    float* agg = (float*)d_ws;

    (void)in_sizes; (void)n_in; (void)out_size; (void)ws_size;

    (void)hipMemsetAsync(agg, 0, (size_t)N_NODES * HIDDEN * sizeof(float), stream);

    edge_msg_kernel<<<N_EDGES / EPB, 256, 0, stream>>>(
        edge_attr, edge_weight, edge_index, z, emb, dp_w, dp_b, agg);

    combine_kernel<<<N_NODES / NPB, 256, 0, stream>>>(
        node_feat, agg, comb_w, comb_b, out);
}
